// HierarchicalQuantizer_61770219651647
// MI455X (gfx1250) — matrix operand order repaired and checked by reading, not compiled, not run on hardware
//
#include <hip/hip_runtime.h>
#include <hip/hip_bf16.h>
#include <math.h>

typedef __attribute__((ext_vector_type(16))) __bf16 v16bf;
typedef __attribute__((ext_vector_type(8)))  __bf16 v8bf;
typedef __attribute__((ext_vector_type(4)))  __bf16 v4bf;
typedef __attribute__((ext_vector_type(8)))  float  v8f;

#define D         256
#define K_CODES   1024
#define NROWS     65536          // 16*4096
#define TILE_M    128
#define TILE_N    64             // codes per chunk (double buffered)
#define NCHUNK    (K_CODES / TILE_N)      // 16
#define OUT_Q     (NROWS * D)    // 16777216 floats of quantized output

// LDS byte map for vq_argmin (dynamic segment starts at LDS offset 0:
// no static __shared__ may be declared in that kernel!)
#define XB_OFF    0
#define XB_SZ     (TILE_M * D * 2)        // 65536
#define EB_OFF    XB_SZ
#define EB_SZ     (TILE_N * D * 2)        // 32768 per buffer
#define SMEM_SZ   (XB_SZ + 2 * EB_SZ)     // 131072

// ---------------------------------------------------------------------------
// Kernel 0: codebook -> bf16 + per-code squared norms
// ---------------------------------------------------------------------------
__global__ __launch_bounds__(256) void prep_codebook(
    const float* __restrict__ emb, __bf16* __restrict__ ebg,
    float* __restrict__ e2g)
{
    __shared__ float red[256];
    const int row = blockIdx.x;          // one code per block
    const int t   = threadIdx.x;         // one column per thread
    float v = emb[row * D + t];
    ebg[row * D + t] = (__bf16)v;
    red[t] = v * v;
    __syncthreads();
    for (int s = 128; s > 0; s >>= 1) {
        if (t < s) red[t] += red[t + s];
        __syncthreads();
    }
    if (t == 0) e2g[row] = red[0];
}

// ---------------------------------------------------------------------------
// Fragment loads from LDS (wave32 WMMA bf16 layouts per CDNA5 ISA 7.12.2)
// ---------------------------------------------------------------------------
// A 16x32 bf16: lane half 0 holds K 0..7 & 16..23, half 1 holds 8..15 & 24..31
__device__ __forceinline__ v16bf load_fragA(const __bf16* base, int lane) {
    const int half = lane >> 4;
    const int r    = lane & 15;
    const __bf16* p = base + r * D;
    v8bf lo = *(const v8bf*)(p + half * 8);        // K 8h .. 8h+7
    v8bf hi = *(const v8bf*)(p + 16 + half * 8);   // K 16+8h .. 23+8h
    v16bf out;
#pragma unroll
    for (int i = 0; i < 8; ++i) { out[i] = lo[i]; out[i + 8] = hi[i]; }
    return out;
}

// B 32x16 bf16: lane = column (code), half 0: K 0..7 & 16..23, half 1: K 8..15 & 24..31
__device__ __forceinline__ v16bf load_fragB(const __bf16* base, int lane) {
    const int half = lane >> 4;
    const int n    = lane & 15;
    const __bf16* p = base + n * D;
    v8bf lo = *(const v8bf*)(p + half * 8);
    v8bf hi = *(const v8bf*)(p + 16 + half * 8);
    v16bf out;
#pragma unroll
    for (int i = 0; i < 8; ++i) { out[i] = lo[i]; out[i + 8] = hi[i]; }
    return out;
}

// ---------------------------------------------------------------------------
// Kernel 1: argmin_n ( ||e_n||^2 - 2 x.e_n )  via bf16 WMMA with
// double-buffered async (ASYNCcnt) codebook staging and software-pipelined
// B-fragment LDS reads.
// ---------------------------------------------------------------------------
__global__ __launch_bounds__(256) void vq_argmin(
    const float*  __restrict__ x,     // [NROWS, D] f32
    const __bf16* __restrict__ ebg,   // [K, D] bf16
    const float*  __restrict__ e2g,   // [K]
    int* __restrict__ idx_out)        // [NROWS]
{
    extern __shared__ char smem_raw[];
    __bf16* xb = (__bf16*)smem_raw;   // at LDS offset 0

    const int tid  = threadIdx.x;
    const int wave = tid >> 5;
    const int lane = tid & 31;
    const int half = lane >> 4;
    const int l16  = lane & 15;
    const int row_base = blockIdx.x * TILE_M;

    // ---- stage x tile -> bf16 in LDS (coalesced float4 reads) ----
    const float4* xg4 = (const float4*)(x + (size_t)row_base * D);
    for (int i = tid; i < TILE_M * D / 4; i += 256) {
        float4 f = xg4[i];
        v4bf b;
        b[0] = (__bf16)f.x; b[1] = (__bf16)f.y;
        b[2] = (__bf16)f.z; b[3] = (__bf16)f.w;
        *((v4bf*)xb + i) = b;
    }

    // ---- async stage of one codebook chunk into LDS (8 x b128 / thread) ----
    auto issue_chunk = [&](int c) {
        const char* src = (const char*)(ebg + (size_t)c * TILE_N * D);
        const unsigned dstbuf = EB_OFF + (unsigned)(c & 1) * EB_SZ;
#pragma unroll
        for (int j = 0; j < 8; ++j) {
            unsigned byte = (unsigned)(tid + j * 256) * 16u;
            unsigned lds  = dstbuf + byte;
            asm volatile("global_load_async_to_lds_b128 %0, %1, %2"
                         :: "v"(lds), "v"(byte), "s"(src) : "memory");
        }
    };

    issue_chunk(0);
    __syncthreads();                       // xb visible to all waves

    // hold this wave's 8 A fragments in registers for the whole sweep
    v16bf afrag[8];
#pragma unroll
    for (int k = 0; k < 8; ++k)
        afrag[k] = load_fragA(xb + wave * 16 * D + k * 32, lane);

    float minval[8];
    int   minidx[8];
#pragma unroll
    for (int v = 0; v < 8; ++v) { minval[v] = 3.4e38f; minidx[v] = 0; }

    for (int c = 0; c < NCHUNK; ++c) {
        if (c + 1 < NCHUNK) {
            issue_chunk(c + 1);            // prefetch into other buffer
            asm volatile("s_wait_asynccnt 8" ::: "memory");  // chunk c done
        } else {
            asm volatile("s_wait_asynccnt 0" ::: "memory");
        }
        __syncthreads();                   // all waves' chunk-c data in LDS

        const __bf16* eb = (const __bf16*)(smem_raw + EB_OFF + (c & 1) * EB_SZ);

        v8f acc[4];
        const v8f zero = {0.f, 0.f, 0.f, 0.f, 0.f, 0.f, 0.f, 0.f};
#pragma unroll
        for (int n = 0; n < 4; ++n) acc[n] = zero;

        // software pipeline: B fragments for step k+1 load while step k's
        // WMMAs run on already-resident registers (no WAR full-drain waits)
        v16bf bcur[4], bnext[4];
#pragma unroll
        for (int n = 0; n < 4; ++n)
            bcur[n] = load_fragB(eb + n * 16 * D, lane);

#pragma unroll
        for (int k = 0; k < 8; ++k) {      // D = 8 * 32
            if (k + 1 < 8) {
#pragma unroll
                for (int n = 0; n < 4; ++n)
                    bnext[n] = load_fragB(eb + n * 16 * D + (k + 1) * 32, lane);
            }
#pragma unroll
            for (int n = 0; n < 4; ++n) {
                acc[n] = __builtin_amdgcn_wmma_f32_16x16x32_bf16(
                    false, afrag[k], false, bcur[n], (short)0, acc[n],
                    false, false);
            }
#pragma unroll
            for (int n = 0; n < 4; ++n) bcur[n] = bnext[n];  // SSA-renamed away
        }

        // distance proxy: ||e||^2 - 2*dot ; track running min
#pragma unroll
        for (int n = 0; n < 4; ++n) {
            const int   code = c * TILE_N + n * 16 + l16;
            const float e2v  = e2g[code];
#pragma unroll
            for (int v = 0; v < 8; ++v) {
                float s = fmaf(-2.0f, acc[n][v], e2v);
                if (s < minval[v]) { minval[v] = s; minidx[v] = code; }
            }
        }
        __syncthreads();                   // readers done before buffer reuse
    }

    // reduce across the 16 lanes of each column group (C layout:
    // lane = column, vgpr v = row v + 8*half)
#pragma unroll
    for (int v = 0; v < 8; ++v) {
        float mv = minval[v];
        int   mi = minidx[v];
#pragma unroll
        for (int off = 1; off < 16; off <<= 1) {
            float ov = __shfl_xor(mv, off, 32);
            int   oi = __shfl_xor(mi, off, 32);
            if (ov < mv || (ov == mv && oi < mi)) { mv = ov; mi = oi; }
        }
        if (l16 == 0) {
            int row = row_base + wave * 16 + v + half * 8;
            idx_out[row] = mi;
        }
    }
}

// ---------------------------------------------------------------------------
// Kernel 2: gather quantized vectors, accumulate loss + code counts
// ---------------------------------------------------------------------------
#define GROWS 16
__global__ __launch_bounds__(256) void gather_loss(
    const float* __restrict__ x, const float* __restrict__ emb,
    const int* __restrict__ idx,
    float* __restrict__ out_q, float* __restrict__ out_idxf,
    unsigned* __restrict__ counts, float* __restrict__ loss_accum)
{
    const int tid = threadIdx.x;
    const int rl  = tid >> 4;   // local row 0..15
    const int cl  = tid & 15;   // column group (16 floats each)
    const int row = blockIdx.x * GROWS + rl;
    const int code = idx[row];

    const float4* xr = (const float4*)(x + (size_t)row * D) + cl * 4;
    const float4* qr = (const float4*)(emb + (size_t)code * D) + cl * 4;
    float4* orow = (float4*)(out_q + (size_t)row * D) + cl * 4;

    float local = 0.f;
#pragma unroll
    for (int j = 0; j < 4; ++j) {
        float4 xv = xr[j], qv = qr[j];
        orow[j] = qv;
        float dx = xv.x - qv.x, dy = xv.y - qv.y;
        float dz = xv.z - qv.z, dw = xv.w - qv.w;
        local += dx * dx + dy * dy + dz * dz + dw * dw;
    }
    if (cl == 0) {
        atomicAdd(&counts[code], 1u);
        out_idxf[row] = (float)code;
    }

    __shared__ float red[256];
    red[tid] = local;
    __syncthreads();
    for (int s = 128; s > 0; s >>= 1) {
        if (tid < s) red[tid] += red[tid + s];
        __syncthreads();
    }
    if (tid == 0) atomicAdd(loss_accum, red[0]);
}

// ---------------------------------------------------------------------------
// Kernel 3: perplexity + final loss scalars
// ---------------------------------------------------------------------------
__global__ __launch_bounds__(256) void finalize(
    const unsigned* __restrict__ counts, const float* __restrict__ loss_accum,
    float* __restrict__ out_loss, float* __restrict__ out_perp)
{
    __shared__ float red[256];
    const int t = threadIdx.x;
    float s = 0.f;
    for (int i = t; i < K_CODES; i += 256) {
        float p = (float)counts[i] * (1.0f / (float)NROWS);
        s += p * logf(p + 1e-10f);
    }
    red[t] = s;
    __syncthreads();
    for (int k = 128; k > 0; k >>= 1) {
        if (t < k) red[t] += red[t + k];
        __syncthreads();
    }
    if (t == 0) {
        *out_perp = expf(-red[0]);
        // codebook + COMMITMENT_COST * commitment = 1.25 * MSE
        *out_loss = 1.25f * (*loss_accum) / ((float)NROWS * (float)D);
    }
}

// ---------------------------------------------------------------------------
extern "C" void kernel_launch(void* const* d_in, const int* in_sizes, int n_in,
                              void* d_out, int out_size, void* d_ws, size_t ws_size,
                              hipStream_t stream) {
    (void)in_sizes; (void)n_in; (void)out_size; (void)ws_size;
    const float* x   = (const float*)d_in[0];   // [16,4096,256] f32
    const float* emb = (const float*)d_in[1];   // [1024,256] f32

    char* ws = (char*)d_ws;
    __bf16*   ebg     = (__bf16*)ws;                       // 524288 B
    float*    e2g     = (float*)(ws + 524288);             //   4096 B
    int*      idxb    = (int*)(ws + 528384);               // 262144 B
    unsigned* counts  = (unsigned*)(ws + 790528);          //   4096 B
    float*    lossacc = (float*)(ws + 794624);             //      4 B

    float* out       = (float*)d_out;
    float* out_q     = out;                 // quantized_st (== quantized fwd)
    float* out_loss  = out + OUT_Q;         // vq_total_loss
    float* out_perp  = out + OUT_Q + 1;     // perplexity
    float* out_idxf  = out + OUT_Q + 2;     // encoding_indices (as float)

    hipMemsetAsync(ws + 790528, 0, 4096 + 4, stream);

    prep_codebook<<<K_CODES, 256, 0, stream>>>(emb, ebg, e2g);

    vq_argmin<<<NROWS / TILE_M, 256, SMEM_SZ, stream>>>(x, ebg, e2g, idxb);

    gather_loss<<<NROWS / GROWS, 256, 0, stream>>>(x, emb, idxb, out_q, out_idxf,
                                                   counts, lossacc);
    finalize<<<1, 256, 0, stream>>>(counts, lossacc, out_loss, out_perp);
}
